// HeteroGAT_ATAEncoder_47828755808350
// MI455X (gfx1250) — matrix operand order repaired and checked by reading, not compiled, not run on hardware
//
#include <hip/hip_runtime.h>
#include <hip/hip_bf16.h>
#include <cstdint>
#include <cstddef>

#define HDIM 256
#define NHEADS 4
#define NSLOPE 0.2f

typedef __attribute__((ext_vector_type(16))) __bf16 v16bf;
typedef __attribute__((ext_vector_type(8)))  float  v8f;

static inline int ceil_div_i(long long a, long long b) { return (int)((a + b - 1) / b); }

// ---------------------------------------------------------------- utilities
__global__ void k_fill(float* __restrict__ p, float v, size_t n) {
  size_t i = (size_t)blockIdx.x * blockDim.x + threadIdx.x;
  if (i < n) p[i] = v;
}

// [M,K] -> [M,Kp] zero-padded copy (row stride Kp)
__global__ void k_pad_rows(const float* __restrict__ X, float* __restrict__ Y,
                           int M, int K, int Kp) {
  long long i = (long long)blockIdx.x * blockDim.x + threadIdx.x;
  if (i >= (long long)M * Kp) return;
  int r = (int)(i / Kp), c = (int)(i - (long long)r * Kp);
  Y[i] = (c < K) ? X[(long long)r * K + c] : 0.0f;
}

// W[K,N] -> WT[N,Kp]  (WT pre-zeroed when Kp > K)
__global__ void k_transpose(const float* __restrict__ W, float* __restrict__ WT,
                            int K, int N, int Kp) {
  int i = blockIdx.x * blockDim.x + threadIdx.x;
  if (i < K * N) {
    int k = i / N, n = i - k * N;
    WT[(size_t)n * Kp + k] = W[i];
  }
}

// ---------------------------------------------------------------- WMMA GEMM
__device__ __forceinline__ v16bf pack_bf16(float4 a, float4 b, float4 c, float4 d) {
  v16bf r;
  r[0]  = (__bf16)a.x; r[1]  = (__bf16)a.y; r[2]  = (__bf16)a.z; r[3]  = (__bf16)a.w;
  r[4]  = (__bf16)b.x; r[5]  = (__bf16)b.y; r[6]  = (__bf16)b.z; r[7]  = (__bf16)b.w;
  r[8]  = (__bf16)c.x; r[9]  = (__bf16)c.y; r[10] = (__bf16)c.z; r[11] = (__bf16)c.w;
  r[12] = (__bf16)d.x; r[13] = (__bf16)d.y; r[14] = (__bf16)d.z; r[15] = (__bf16)d.w;
  return r;
}

// C[M,N] = A[M,K] @ B[K,N] (+ bias). B supplied transposed+padded: BT[N,K].
// Requirements: K % 32 == 0, A rows are [M,K] with 16B-aligned base, N % 32 == 0.
// One wave computes a 16x32 tile (two v_wmma_f32_16x16x32_bf16 per K-step,
// sharing the A fragment); 8 waves/block stack along M (128-row block tile).
// No branches inside the K loop: out-of-range rows are clamped for loads and
// masked only at the store, so EXEC is all-1s for every WMMA.
__global__ void k_gemm_wmma(const float* __restrict__ A, const float* __restrict__ BT,
                            const float* __restrict__ bias, float* __restrict__ C,
                            int M, int N, int K, int useBias) {
  const int lane = threadIdx.x & 31;
  const int wave = threadIdx.x >> 5;
  const int row0 = (blockIdx.y * 8 + wave) * 16;
  const int col0 = blockIdx.x * 32;
  if (row0 >= M) return;                         // uniform across the wave
  const int half = lane >> 4;                    // 0 | 1
  const int lm   = lane & 15;
  const int arow = row0 + lm;
  const bool aok = arow < M;
  const float4* ap4 = (const float4*)(A  + (size_t)(aok ? arow : 0) * K);
  const float4* bp4 = (const float4*)(BT + (size_t)(col0 + lm) * K);
  const float4* cp4 = (const float4*)(BT + (size_t)(col0 + 16 + lm) * K);
  const int kb = half * 8;                       // A-fragment K base per lane half
  const int kB = half * 8;                       // B-fragment K base per lane half

  v8f acc0 = {}, acc1 = {};
  for (int k0 = 0; k0 < K; k0 += 32) {
    __builtin_prefetch((const float*)ap4 + k0 + 128, 0, 1);
    // A 16x32 bf16 layout: elems 0..7 -> K=kb+0..7 ; elems 8..15 -> K=16+kb+0..7
    float4 a0 = ap4[(k0 + kb) >> 2];
    float4 a1 = ap4[((k0 + kb) >> 2) + 1];
    float4 a2 = ap4[(k0 + 16 + kb) >> 2];
    float4 a3 = ap4[((k0 + 16 + kb) >> 2) + 1];
    // B 32x16 bf16 layout: lane holds one column; elems 0..7 -> K=kB+0..7 ; elems 8..15 -> K=16+kB+0..7
    int bi = (k0 + kB) >> 2;
    float4 b0 = bp4[bi], b1 = bp4[bi + 1], b2 = bp4[bi + 4], b3 = bp4[bi + 5];
    float4 c0 = cp4[bi], c1 = cp4[bi + 1], c2 = cp4[bi + 4], c3 = cp4[bi + 5];
    v16bf af  = pack_bf16(a0, a1, a2, a3);
    v16bf bf0 = pack_bf16(b0, b1, b2, b3);
    v16bf bf1 = pack_bf16(c0, c1, c2, c3);
    acc0 = __builtin_amdgcn_wmma_f32_16x16x32_bf16(false, af, false, bf0,
                                                   (short)0, acc0, false, false);
    acc1 = __builtin_amdgcn_wmma_f32_16x16x32_bf16(false, af, false, bf1,
                                                   (short)0, acc1, false, false);
  }
  const int n0 = col0 + lm;
  const int n1 = col0 + 16 + lm;
  const float badd0 = useBias ? bias[n0] : 0.0f;
  const float badd1 = useBias ? bias[n1] : 0.0f;
#pragma unroll
  for (int j = 0; j < 8; ++j) {
    int m = row0 + half * 8 + j;                 // C/D layout: row = half*8 + j
    if (m < M) {
      C[(size_t)m * N + n0] = acc0[j] + badd0;
      C[(size_t)m * N + n1] = acc1[j] + badd1;
    }
  }
}

// ---------------------------------------------------------------- attention
__global__ void k_node_logits(const float* __restrict__ h, const float* __restrict__ avec,
                              float* __restrict__ out, int nn) {
  int i = blockIdx.x * blockDim.x + threadIdx.x;
  if (i >= nn * NHEADS) return;
  int node = i >> 2, head = i & 3;
  const float* hp = h + (size_t)node * HDIM + head * 64;
  const float* a  = avec + head * 64;
  float s = 0.0f;
#pragma unroll 8
  for (int j = 0; j < 64; ++j) s += hp[j] * a[j];
  out[i] = s;
}

__device__ __forceinline__ void atomicMaxF(float* addr, float val) {
  if (val >= 0.0f) atomicMax((int*)addr, __float_as_int(val));
  else             atomicMin((unsigned int*)addr, (unsigned int)__float_as_int(val));
}

__device__ __forceinline__ float leaky(float x) { return x >= 0.0f ? x : NSLOPE * x; }

__global__ void k_edge_max(const float* __restrict__ als, const float* __restrict__ ald,
                           const int* __restrict__ src, const int* __restrict__ dst,
                           float* __restrict__ mx, int E) {
  int i = blockIdx.x * blockDim.x + threadIdx.x;
  if (i >= E * NHEADS) return;
  int e = i >> 2, h = i & 3;
  int s = src[e], d = dst[e];
  float al = leaky(als[s * NHEADS + h] + ald[d * NHEADS + h]);
  atomicMaxF(&mx[d * NHEADS + h], al);
}

__global__ void k_edge_exp(const float* __restrict__ als, const float* __restrict__ ald,
                           const int* __restrict__ src, const int* __restrict__ dst,
                           const float* __restrict__ mx, float* __restrict__ den,
                           float* __restrict__ ex, int E) {
  int i = blockIdx.x * blockDim.x + threadIdx.x;
  if (i >= E * NHEADS) return;
  int e = i >> 2, h = i & 3;
  int s = src[e], d = dst[e];
  float al = leaky(als[s * NHEADS + h] + ald[d * NHEADS + h]);
  float v = __expf(al - mx[d * NHEADS + h]);
  ex[i] = v;
  atomicAdd(&den[d * NHEADS + h], v);
}

// one wave per edge: 8 features per lane, alpha uniform within the 8 (64 | 8)
__global__ void k_edge_scatter(const float* __restrict__ hs, const int* __restrict__ src,
                               const int* __restrict__ dst, const float* __restrict__ ex,
                               const float* __restrict__ den, float* __restrict__ acc, int E) {
  int t = blockIdx.x * blockDim.x + threadIdx.x;
  int e = t >> 5, lane = t & 31;
  if (e >= E) return;
  int s = src[e], d = dst[e];
  int f0 = lane * 8;
  int head = f0 >> 6;
  float alpha = ex[e * NHEADS + head] / den[d * NHEADS + head];
  const float* hp = hs + (size_t)s * HDIM + f0;
  float* apo = acc + (size_t)d * HDIM + f0;
#pragma unroll
  for (int j = 0; j < 8; ++j) atomicAdd(&apo[j], hp[j] * alpha);
}

// ---------------------------------------------------------------- LN+res+ELU
__global__ void k_ln_elu(const float* __restrict__ acc, const float* __restrict__ bias,
                         const float* __restrict__ g, const float* __restrict__ b,
                         float* __restrict__ h, int doRes) {
  __shared__ float red[HDIM];
  int node = blockIdx.x, f = threadIdx.x;
  float x = acc[(size_t)node * HDIM + f] + bias[f];
  red[f] = x; __syncthreads();
  for (int s = HDIM / 2; s > 0; s >>= 1) { if (f < s) red[f] += red[f + s]; __syncthreads(); }
  float mu = red[0] * (1.0f / HDIM); __syncthreads();
  float dx = x - mu;
  red[f] = dx * dx; __syncthreads();
  for (int s = HDIM / 2; s > 0; s >>= 1) { if (f < s) red[f] += red[f + s]; __syncthreads(); }
  float var = red[0] * (1.0f / HDIM);
  float y = dx * rsqrtf(var + 1e-5f) * g[f] + b[f];
  if (doRes) y += h[(size_t)node * HDIM + f];
  h[(size_t)node * HDIM + f] = (y > 0.0f) ? y : (__expf(y) - 1.0f);   // ELU
}

// ---------------------------------------------------------------- pooling
__global__ void k_pool_scatter(const float* __restrict__ h_addr, const int* __restrict__ aidx,
                               const int* __restrict__ tidx, float* __restrict__ sum,
                               float* __restrict__ cnt, int E) {
  int t = blockIdx.x * blockDim.x + threadIdx.x;
  int e = t >> 5, lane = t & 31;
  if (e >= E) return;
  int a = aidx[e], d = tidx[e];
  int f0 = lane * 8;
  const float* hp = h_addr + (size_t)a * HDIM + f0;
  float* sp = sum + (size_t)d * HDIM + f0;
#pragma unroll
  for (int j = 0; j < 8; ++j) atomicAdd(&sp[j], hp[j]);
  if (lane == 0) atomicAdd(&cnt[d], 1.0f);
}

__global__ void k_pool_div(const float* __restrict__ sum, const float* __restrict__ cnt,
                           float* __restrict__ out, int n) {
  long long i = (long long)blockIdx.x * blockDim.x + threadIdx.x;
  if (i < (long long)n * HDIM) {
    int node = (int)(i / HDIM);
    out[i] = sum[i] / fmaxf(cnt[node], 1.0f);
  }
}

// ---------------------------------------------------------------- host driver
extern "C" void kernel_launch(void* const* d_in, const int* in_sizes, int n_in,
                              void* d_out, int out_size, void* d_ws, size_t ws_size,
                              hipStream_t stream) {
  (void)in_sizes; (void)n_in; (void)out_size; (void)ws_size;
  const int N_TX = 100000, N_ADDR = 200000, E = 400000;
  const int F_TX = 165, F_TX_P = 192, F_ADDR = 64, H = 64;

  const float* x_tx      = (const float*)d_in[0];
  const float* x_addr    = (const float*)d_in[1];
  const int*   addr_at   = (const int*)d_in[2];
  const int*   tx_at     = (const int*)d_in[3];
  const int*   tx_ta     = (const int*)d_in[4];
  const int*   addr_ta   = (const int*)d_in[5];
  const float* w_in_tx   = (const float*)d_in[6];
  const float* b_in_tx   = (const float*)d_in[7];
  const float* w_in_addr = (const float*)d_in[8];
  const float* b_in_addr = (const float*)d_in[9];
  const float* Wat[2]  = { (const float*)d_in[10], (const float*)d_in[18] };
  const float* ASat[2] = { (const float*)d_in[11], (const float*)d_in[19] };
  const float* ADat[2] = { (const float*)d_in[12], (const float*)d_in[20] };
  const float* Bat[2]  = { (const float*)d_in[13], (const float*)d_in[21] };
  const float* Wta[2]  = { (const float*)d_in[14], (const float*)d_in[22] };
  const float* ASta[2] = { (const float*)d_in[15], (const float*)d_in[23] };
  const float* ADta[2] = { (const float*)d_in[16], (const float*)d_in[24] };
  const float* Bta[2]  = { (const float*)d_in[17], (const float*)d_in[25] };
  const float* ln_g_tx   = (const float*)d_in[26];
  const float* ln_b_tx   = (const float*)d_in[27];
  const float* ln_g_addr = (const float*)d_in[28];
  const float* ln_b_addr = (const float*)d_in[29];
  const float* w_out     = (const float*)d_in[30];
  const float* b_out     = (const float*)d_in[31];

  // ---- workspace carve-up (floats; all sizes multiples of 4 -> 16B alignment kept)
  float* ws = (float*)d_ws;
  size_t off = 0;
  auto alloc = [&](size_t n) { float* p = ws + off; off += n; return p; };
  float* h_tx     = alloc((size_t)N_TX   * HDIM);
  float* h_addr   = alloc((size_t)N_ADDR * HDIM);
  float* hs       = alloc((size_t)N_ADDR * HDIM);   // projected src feats (max count)
  float* hd       = alloc((size_t)N_ADDR * HDIM);   // projected dst feats (max count)
  float* acc_tx   = alloc((size_t)N_TX   * HDIM);   // reused as pool-sum at the end
  float* acc_addr = alloc((size_t)N_ADDR * HDIM);
  float* als      = alloc((size_t)N_ADDR * NHEADS);
  float* ald      = alloc((size_t)N_ADDR * NHEADS);
  float* mx       = alloc((size_t)N_ADDR * NHEADS);
  float* den      = alloc((size_t)N_ADDR * NHEADS);
  float* ex       = alloc((size_t)E * NHEADS);
  float* wT       = alloc((size_t)HDIM * HDIM);
  float* cnt      = alloc((size_t)N_TX);
  float* xpad     = alloc((size_t)N_TX * F_TX_P);   // K=165 -> 192, zero-padded

  // gemm: K need not be padded by caller; we pad weights (and assume A stride = Kp)
  auto gemm = [&](const float* A, const float* W, const float* bias, float* C,
                  int M, int N, int K, int useBias) {
    int Kp = (K + 31) & ~31;
    if (Kp != K)
      k_fill<<<ceil_div_i((long long)N * Kp, 256), 256, 0, stream>>>(wT, 0.0f, (size_t)N * Kp);
    k_transpose<<<ceil_div_i((long long)K * N, 256), 256, 0, stream>>>(W, wT, K, N, Kp);
    dim3 g(N / 32, ceil_div_i(M, 128));
    k_gemm_wmma<<<g, 256, 0, stream>>>(A, wT, bias, C, M, N, Kp, useBias);
  };

  auto gat = [&](const float* hsrc, const float* hdst, int Nsrc, int Ndst, int din,
                 const int* srcIdx, const int* dstIdx,
                 const float* W, const float* a_s, const float* a_d, float* accOut) {
    gemm(hsrc, W, nullptr, hs, Nsrc, HDIM, din, 0);
    gemm(hdst, W, nullptr, hd, Ndst, HDIM, din, 0);
    k_node_logits<<<ceil_div_i((long long)Nsrc * NHEADS, 256), 256, 0, stream>>>(hs, a_s, als, Nsrc);
    k_node_logits<<<ceil_div_i((long long)Ndst * NHEADS, 256), 256, 0, stream>>>(hd, a_d, ald, Ndst);
    k_fill<<<ceil_div_i((long long)Ndst * NHEADS, 256), 256, 0, stream>>>(mx, -INFINITY, (size_t)Ndst * NHEADS);
    k_fill<<<ceil_div_i((long long)Ndst * NHEADS, 256), 256, 0, stream>>>(den, 0.0f, (size_t)Ndst * NHEADS);
    k_fill<<<ceil_div_i((long long)Ndst * HDIM, 256), 256, 0, stream>>>(accOut, 0.0f, (size_t)Ndst * HDIM);
    k_edge_max<<<ceil_div_i((long long)E * NHEADS, 256), 256, 0, stream>>>(als, ald, srcIdx, dstIdx, mx, E);
    k_edge_exp<<<ceil_div_i((long long)E * NHEADS, 256), 256, 0, stream>>>(als, ald, srcIdx, dstIdx, mx, den, ex, E);
    k_edge_scatter<<<ceil_div_i((long long)E * 32, 256), 256, 0, stream>>>(hs, srcIdx, dstIdx, ex, den, accOut, E);
  };

  // ---- input projections (x_tx padded 165->192 so GEMM rows are aligned, K%32==0)
  k_pad_rows<<<ceil_div_i((long long)N_TX * F_TX_P, 256), 256, 0, stream>>>(x_tx, xpad, N_TX, F_TX, F_TX_P);
  gemm(xpad,   w_in_tx,   b_in_tx,   h_tx,   N_TX,   H, F_TX,   1);
  gemm(x_addr, w_in_addr, b_in_addr, h_addr, N_ADDR, H, F_ADDR, 1);

  // ---- two GAT layers
  int din = H;
  for (int L = 0; L < 2; ++L) {
    gat(h_addr, h_tx, N_ADDR, N_TX, din, addr_at, tx_at, Wat[L], ASat[L], ADat[L], acc_tx);
    gat(h_tx, h_addr, N_TX, N_ADDR, din, tx_ta, addr_ta, Wta[L], ASta[L], ADta[L], acc_addr);
    int doRes = (din == HDIM) ? 1 : 0;     // residual only when dims match (layer 1)
    k_ln_elu<<<N_TX,   HDIM, 0, stream>>>(acc_tx,   Bat[L], ln_g_tx,   ln_b_tx,   h_tx,   doRes);
    k_ln_elu<<<N_ADDR, HDIM, 0, stream>>>(acc_addr, Bta[L], ln_g_addr, ln_b_addr, h_addr, doRes);
    din = HDIM;
  }

  // ---- scatter-mean pooling addr -> tx, then output projection (WMMA)
  k_fill<<<ceil_div_i((long long)N_TX * HDIM, 256), 256, 0, stream>>>(acc_tx, 0.0f, (size_t)N_TX * HDIM);
  k_fill<<<ceil_div_i((long long)N_TX, 256), 256, 0, stream>>>(cnt, 0.0f, (size_t)N_TX);
  k_pool_scatter<<<ceil_div_i((long long)E * 32, 256), 256, 0, stream>>>(h_addr, addr_at, tx_at, acc_tx, cnt, E);
  k_pool_div<<<ceil_div_i((long long)N_TX * HDIM, 256), 256, 0, stream>>>(acc_tx, cnt, hd, N_TX);
  gemm(hd, w_out, b_out, (float*)d_out, N_TX, HDIM, HDIM, 1);
}
